// AdvancedGNN_12317966205294
// MI455X (gfx1250) — matrix operand order repaired and checked by reading, not compiled, not run on hardware
//
#include <hip/hip_runtime.h>
#include <math.h>

// ---------------------------------------------------------------------------
// CDNA5 / gfx1250 WMMA helpers (wave32).
// ---------------------------------------------------------------------------
#define DEVI __device__ __forceinline__

typedef __attribute__((ext_vector_type(16))) _Float16 v16h;
typedef __attribute__((ext_vector_type(8)))  _Float16 v8h;
typedef __attribute__((ext_vector_type(8)))  float    v8f;

// A-fragment (16x32 f16, row-major LDS buffer with leading dim `ld`).
// lanes 0-15 -> M=lane, K=0..15; lanes 16-31 -> K=16..31 (matches packed B).
DEVI v16h load_a_frag(const _Float16* buf, int ld, int kb, int lane) {
  const _Float16* p = buf + (lane & 15) * ld + kb * 32 + ((lane >> 4) << 4);
  union { v16h v; v8h h[2]; } u;
  u.h[0] = *(const v8h*)(p);        // K = base + 0..7   (or 16..23)
  u.h[1] = *(const v8h*)(p + 8);    // K = base + 8..15  (or 24..31)
  return u.v;
}

// B-fragment from pre-packed global weights: per (kb,nb) tile, 32 lanes x 16 f16,
// lane holds N = lane%16, K = j + 16*(lane>=16).
DEVI v16h load_b_frag(const _Float16* pk, int kb, int nb, int NB, int lane) {
  return *(const v16h*)(pk + ((size_t)((kb * NB + nb) * 32 + lane) << 4));
}

DEVI v8f wmma_f16(v16h a, v16h b, v8f c) {
  // 8 args: (neg_a, A, neg_b, B, c_mod, C, reuse_a, reuse_b)
  return __builtin_amdgcn_wmma_f32_16x16x32_f16(false, a, false, b, (short)0, c,
                                                false, false);
}

DEVI float lrelu(float x) { return x > 0.f ? x : 0.1f * x; }
DEVI float geluf(float x) { return 0.5f * x * (1.f + erff(x * 0.70710678118f)); }
DEVI float actf(float x, int g) { return g ? geluf(x) : lrelu(x); }
DEVI float sigmf(float x) { return 1.f / (1.f + __expf(-x)); }

// ---------------------------------------------------------------------------
// Weight packing: fp32 W[O][I] -> f16 B-fragment tiles, K padded to KB*32.
// kOff shifts the logical K origin (used for emb "pow" occupying K=16..31).
// ---------------------------------------------------------------------------
__global__ void pack_w(const float* __restrict__ W, int O, int I, int kOff,
                       int KB, int NB, _Float16* __restrict__ dst) {
  int idx = blockIdx.x * blockDim.x + threadIdx.x;
  int total = KB * NB * 512;
  if (idx >= total) return;
  int j    = idx & 15;
  int lane = (idx >> 4) & 31;
  int tile = idx >> 9;
  int nb = tile % NB, kb = tile / NB;
  int n = nb * 16 + (lane & 15);
  int k = kb * 32 + j + ((lane >> 4) << 4) - kOff;
  float v = (n < O && k >= 0 && k < I) ? W[(size_t)n * I + k] : 0.0f;
  dst[idx] = (_Float16)v;
}

// ---------------------------------------------------------------------------
// Embedding: h = comb([lin(x) | pow(x^2)]) ; 16 nodes per wave, 4 waves/block.
// Also writes the f16 mirror hf consumed by the edge-gather DMA.
// ---------------------------------------------------------------------------
struct EmbArgs {
  const float* x; float* h; _Float16* hf;
  const _Float16 *pklin, *pkpow, *pkcomb;
  const float *lin_b, *pow_b, *comb_b;
  int N;
};

#define EMB_WB 9216
__global__ void __launch_bounds__(128) embed_kernel(EmbArgs A) {
  extern __shared__ char lds[];
  const int lane = threadIdx.x & 31, wave = threadIdx.x >> 5;
  char* wl = lds + (size_t)wave * EMB_WB;
  _Float16* xb = (_Float16*)(wl);          // 16 x 32   [x | x^2]
  _Float16* tb = (_Float16*)(wl + 1024);   // 16 x 256  [t1 | t2]
  const int n0 = (blockIdx.x * 4 + wave) * 16;

  for (int idx = lane; idx < 512; idx += 32) {
    int m = idx >> 5, c = idx & 31, node = n0 + m;
    float v = 0.f;
    if (node < A.N) {
      float xv = A.x[(size_t)node * 16 + (c & 15)];
      v = (c < 16) ? xv : xv * xv;
    }
    xb[idx] = (_Float16)v;
  }
  __syncthreads();

  for (int nb = 0; nb < 16; ++nb) {
    v8f c1 = {};
    v16h a = load_a_frag(xb, 32, 0, lane);
    const _Float16* pk = (nb < 8) ? A.pklin : A.pkpow;
    c1 = wmma_f16(a, load_b_frag(pk, 0, nb & 7, 8, lane), c1);
    int ncol = nb * 16 + (lane & 15);
    float bb = (nb < 8) ? A.lin_b[ncol] : A.pow_b[ncol - 128];
    int rbase = (lane >> 4) << 3;
    for (int r = 0; r < 8; ++r) tb[(rbase + r) * 256 + ncol] = (_Float16)(c1[r] + bb);
  }
  __syncthreads();

  for (int nb = 0; nb < 8; ++nb) {
    v8f c1 = {};
    for (int kb = 0; kb < 8; ++kb)
      c1 = wmma_f16(load_a_frag(tb, 256, kb, lane),
                    load_b_frag(A.pkcomb, kb, nb, 8, lane), c1);
    int n = nb * 16 + (lane & 15);
    float bb = A.comb_b[n];
    int rbase = (lane >> 4) << 3;
    for (int r = 0; r < 8; ++r) {
      int node = n0 + rbase + r;
      if (node < A.N) {
        float hv = c1[r] + bb;
        size_t off = (size_t)node * 128 + n;
        A.h[off]  = hv;
        A.hf[off] = (_Float16)hv;
      }
    }
  }
}

// ---------------------------------------------------------------------------
// Edge message kernel: 16 edges per wave, 4 waves/block, dynamic LDS.
//   mi = [hf[dst] | hf[src] | ea | pad] (16x288 f16); node rows are pulled with
//   GLOBAL_LOAD_ASYNC_TO_LDS_B128 (ASYNCcnt-tracked DMA, no VGPR round-trip).
//   m1 = m1_2(LN(act(m1_1(mi))));  m2 = m2_2(LN(gelu(m2_1(mi))))
//   aw = softmax(att([m1|m2]));  msg = (aw0*m1 + aw1*m2) * scale
//   aggr[dst] += msg  (global_atomic_add_f32)
// ---------------------------------------------------------------------------
struct EdgeArgs {
  const _Float16* hf; const int* ei; const float* ea;
  const _Float16 *pk11, *pk21, *pk12, *pk22;
  const float *b11, *b21, *b12, *b22;
  const float *ln1g, *ln1b, *ln2g, *ln2b;
  const float *attw, *attb, *scale;
  float* aggr;
  int E, Etot, act;
};

#define EDGE_WB 18432
__global__ void __launch_bounds__(128) edge_msg_kernel(EdgeArgs A) {
  extern __shared__ char lds[];
  const int lane = threadIdx.x & 31, wave = threadIdx.x >> 5;
  char* wl = lds + (size_t)wave * EDGE_WB;
  _Float16* mi    = (_Float16*)(wl);           // 16 x 288
  _Float16* m1buf = (_Float16*)(wl + 9216);    // 16 x 128
  _Float16* m2buf = (_Float16*)(wl + 13312);   // 16 x 128
  char* st = wl + 17408;
  int*   sdst  = (int*)(st);            // 16
  int*   ssrc  = (int*)(st + 64);       // 16
  float* smean = (float*)(st + 128);    // 32  (0-15 m1 rows, 16-31 m2 rows)
  float* srstd = (float*)(st + 256);    // 32
  float* slog  = (float*)(st + 384);    // 16 x 2
  float* saw   = (float*)(st + 512);    // 16 x 2
  float* ssc   = (float*)(st + 640);    // 16
  _Float16* m1fin = mi;                 // reuse mi region after GEMM1
  _Float16* m2fin = mi + 2048;

  const int e0 = (blockIdx.x * 4 + wave) * 16;

  // Phase 0: endpoints + self-loop handling (self-loops appended after E).
  if (lane < 16) {
    int e = e0 + lane, s = 0, d = 0;
    if (e < A.Etot) {
      if (e < A.E) { s = A.ei[e]; d = A.ei[A.E + e]; }
      else         { s = d = e - A.E; }
    }
    sdst[lane] = d; ssrc[lane] = s;
    // is_self == (x_i == x_j); src==dst is the effective condition.
    ssc[lane] = (s == d) ? 1.0f : A.scale[0];
  }
  __syncthreads();

  // Phase 1a: async DMA gather of hf rows straight into LDS (cols 0..255).
  // 16 rows x 2 segments(dst/src) x 16 chunks of 16B = 512 B128 transfers/wave.
  {
    const unsigned miBase = (unsigned)((char*)mi - lds);  // dyn-LDS base = 0
    for (int t = lane; t < 512; t += 32) {
      int m   = t >> 5;         // edge row in tile
      int seg = (t >> 4) & 1;   // 0: hf[dst] -> cols 0..127, 1: hf[src] -> 128..255
      int ch  = t & 15;         // 16-byte chunk
      int node = seg ? ssrc[m] : sdst[m];
      unsigned ldsoff = miBase + (unsigned)(m * 576 + seg * 256 + ch * 16);
      const char* gaddr = (const char*)A.hf + ((size_t)node * 256 + ch * 16);
      asm volatile("global_load_async_to_lds_b128 %0, %1, off"
                   :: "v"(ldsoff), "v"(gaddr) : "memory");
    }
    // Phase 1b: edge-attr + zero padding columns (256..287) via plain ds stores.
    for (int t = lane; t < 512; t += 32) {
      int m = t >> 5, c = 256 + (t & 31), e = e0 + m;
      float v = 0.f;
      if (e < A.Etot && c < 260)
        v = (e < A.E) ? A.ea[(size_t)e * 4 + (c - 256)] : ((c == 256) ? 1.f : 0.f);
      mi[m * 288 + c] = (_Float16)v;
    }
    asm volatile("s_wait_asynccnt 0x0" ::: "memory");
  }
  __syncthreads();

  // Phase 2: GEMM1 (K=288 -> 9 kblocks), bias + activation -> LDS.
  for (int nb = 0; nb < 8; ++nb) {
    if (nb + 1 < 8) {  // prefetch next weight tiles (global_prefetch_b8)
      __builtin_prefetch(A.pk11 + ((size_t)(nb + 1) * 32 << 4), 0, 1);
      __builtin_prefetch(A.pk21 + ((size_t)(nb + 1) * 32 << 4), 0, 1);
    }
    v8f c1 = {}, c2 = {};
    for (int kb = 0; kb < 9; ++kb) {
      v16h a = load_a_frag(mi, 288, kb, lane);
      c1 = wmma_f16(a, load_b_frag(A.pk11, kb, nb, 8, lane), c1);
      c2 = wmma_f16(a, load_b_frag(A.pk21, kb, nb, 8, lane), c2);
    }
    int n = nb * 16 + (lane & 15);
    float bb1 = A.b11[n], bb2 = A.b21[n];
    int rbase = (lane >> 4) << 3;
    for (int r = 0; r < 8; ++r) {
      int m = rbase + r;
      m1buf[m * 128 + n] = (_Float16)actf(c1[r] + bb1, A.act);
      m2buf[m * 128 + n] = (_Float16)geluf(c2[r] + bb2);
    }
  }
  __syncthreads();

  // Phase 3: LayerNorm stats (lanes 0-15: m1 rows, 16-31: m2 rows) + normalize.
  {
    const _Float16* buf = (lane < 16) ? m1buf : m2buf;
    int m = lane & 15;
    float s = 0.f, ss = 0.f;
    for (int c = 0; c < 128; ++c) { float v = (float)buf[m * 128 + c]; s += v; ss += v * v; }
    float mean = s * (1.f / 128.f);
    float var  = ss * (1.f / 128.f) - mean * mean;
    smean[lane] = mean;
    srstd[lane] = rsqrtf(var + 1e-5f);
  }
  if (lane < 16) { slog[lane * 2] = A.attb[0]; slog[lane * 2 + 1] = A.attb[1]; }
  __syncthreads();
  for (int idx = lane; idx < 2048; idx += 32) {
    int m = idx >> 7, c = idx & 127;
    float v1 = ((float)m1buf[idx] - smean[m])      * srstd[m]      * A.ln1g[c] + A.ln1b[c];
    float v2 = ((float)m2buf[idx] - smean[16 + m]) * srstd[16 + m] * A.ln2g[c] + A.ln2b[c];
    m1buf[idx] = (_Float16)v1;
    m2buf[idx] = (_Float16)v2;
  }
  __syncthreads();

  // Phase 4: GEMM2 (K=128) + attention-logit partials (ds_add_f32).
  for (int nb = 0; nb < 8; ++nb) {
    v8f c1 = {}, c2 = {};
    for (int kb = 0; kb < 4; ++kb) {
      c1 = wmma_f16(load_a_frag(m1buf, 128, kb, lane),
                    load_b_frag(A.pk12, kb, nb, 8, lane), c1);
      c2 = wmma_f16(load_a_frag(m2buf, 128, kb, lane),
                    load_b_frag(A.pk22, kb, nb, 8, lane), c2);
    }
    int n = nb * 16 + (lane & 15);
    float bb1 = A.b12[n], bb2 = A.b22[n];
    // att.w is [2][256]; cols 0-127 weight m1, 128-255 weight m2.
    float w10 = A.attw[n],       w20 = A.attw[128 + n];
    float w11 = A.attw[256 + n], w21 = A.attw[384 + n];
    int rbase = (lane >> 4) << 3;
    for (int r = 0; r < 8; ++r) {
      int m = rbase + r;
      float v1 = c1[r] + bb1, v2 = c2[r] + bb2;
      m1fin[m * 128 + n] = (_Float16)v1;
      m2fin[m * 128 + n] = (_Float16)v2;
      atomicAdd(&slog[m * 2 + 0], w10 * v1 + w20 * v2);
      atomicAdd(&slog[m * 2 + 1], w11 * v1 + w21 * v2);
    }
  }
  __syncthreads();

  // Phase 5: 2-way softmax per edge.
  if (lane < 16) {
    float l0 = slog[lane * 2], l1 = slog[lane * 2 + 1];
    float mx = fmaxf(l0, l1);
    float p0 = __expf(l0 - mx), p1 = __expf(l1 - mx);
    float inv = 1.f / (p0 + p1);
    saw[lane * 2] = p0 * inv; saw[lane * 2 + 1] = p1 * inv;
  }
  __syncthreads();

  // Phase 6: blend, scale, scatter-add into aggr[dst].
  for (int idx = lane; idx < 2048; idx += 32) {
    int m = idx >> 7, c = idx & 127, e = e0 + m;
    if (e < A.Etot) {
      float v = saw[m * 2] * (float)m1fin[idx] + saw[m * 2 + 1] * (float)m2fin[idx];
      atomicAdd(&A.aggr[(size_t)sdst[m] * 128 + c], v * ssc[m]);
    }
  }
}

// ---------------------------------------------------------------------------
// Node update: ui=[aggr|h]; g=sigmoid(gate(ui)); u1=LN(act(u1(ui)));
// u2=h+act(u2(u1)); hn=h*(1-g)+u2*g; out=LN(hn); h = (parity? h+out : out).
// Also refreshes the f16 mirror hf for the next layer's edge gather.
// ---------------------------------------------------------------------------
struct NodeArgs {
  const float* aggr; float* h; _Float16* hf; float* lo;
  const _Float16 *pkgate, *pku1, *pku2;
  const float *gate_b, *u1_b, *u2_b;
  const float *u1lng, *u1lnb, *normg, *normb;
  int N, act, parity;
};

#define NODE_WB 16640
__global__ void __launch_bounds__(128) node_update_kernel(NodeArgs A) {
  extern __shared__ char lds[];
  const int lane = threadIdx.x & 31, wave = threadIdx.x >> 5;
  char* wl = lds + (size_t)wave * NODE_WB;
  _Float16* ui    = (_Float16*)(wl);           // 16 x 256
  _Float16* u1buf = (_Float16*)(wl + 8192);    // 16 x 128
  _Float16* gbuf  = (_Float16*)(wl + 12288);   // 16 x 128
  float* smean = (float*)(wl + 16384);         // 16
  float* srstd = (float*)(wl + 16448);         // 16
  float* hn = (float*)wl;                      // 16 x 128 f32 (reuses ui)
  const int n0 = (blockIdx.x * 4 + wave) * 16;

  for (int idx = lane; idx < 4096; idx += 32) {
    int m = idx >> 8, c = idx & 255, node = n0 + m;
    float v = 0.f;
    if (node < A.N)
      v = (c < 128) ? A.aggr[(size_t)node * 128 + c]
                    : A.h[(size_t)node * 128 + (c - 128)];
    ui[idx] = (_Float16)v;
  }
  __syncthreads();

  for (int nb = 0; nb < 8; ++nb) {
    v8f cg = {}, cu = {};
    for (int kb = 0; kb < 8; ++kb) {
      v16h a = load_a_frag(ui, 256, kb, lane);
      cg = wmma_f16(a, load_b_frag(A.pkgate, kb, nb, 8, lane), cg);
      cu = wmma_f16(a, load_b_frag(A.pku1,   kb, nb, 8, lane), cu);
    }
    int n = nb * 16 + (lane & 15);
    float bg = A.gate_b[n], bu = A.u1_b[n];
    int rbase = (lane >> 4) << 3;
    for (int r = 0; r < 8; ++r) {
      int m = rbase + r;
      gbuf[m * 128 + n]  = (_Float16)sigmf(cg[r] + bg);
      u1buf[m * 128 + n] = (_Float16)actf(cu[r] + bu, A.act);
    }
  }
  __syncthreads();

  if (lane < 16) {
    float s = 0.f, ss = 0.f;
    for (int c = 0; c < 128; ++c) { float v = (float)u1buf[lane * 128 + c]; s += v; ss += v * v; }
    float mean = s * (1.f / 128.f);
    float var  = ss * (1.f / 128.f) - mean * mean;
    smean[lane] = mean; srstd[lane] = rsqrtf(var + 1e-5f);
  }
  __syncthreads();
  for (int idx = lane; idx < 2048; idx += 32) {
    int m = idx >> 7, c = idx & 127;
    float v = ((float)u1buf[idx] - smean[m]) * srstd[m] * A.u1lng[c] + A.u1lnb[c];
    u1buf[idx] = (_Float16)v;
  }
  __syncthreads();

  for (int nb = 0; nb < 8; ++nb) {
    v8f c2 = {};
    for (int kb = 0; kb < 4; ++kb)
      c2 = wmma_f16(load_a_frag(u1buf, 128, kb, lane),
                    load_b_frag(A.pku2, kb, nb, 8, lane), c2);
    int n = nb * 16 + (lane & 15);
    float bb = A.u2_b[n];
    int rbase = (lane >> 4) << 3;
    for (int r = 0; r < 8; ++r) {
      int m = rbase + r, node = n0 + m;
      float hv  = (node < A.N) ? A.h[(size_t)node * 128 + n] : 0.f;
      float u2v = hv + actf(c2[r] + bb, A.act);
      float g   = (float)gbuf[m * 128 + n];
      hn[m * 128 + n] = hv * (1.f - g) + u2v * g;  // hn reuses ui region
    }
  }
  __syncthreads();

  if (lane < 16) {
    float s = 0.f, ss = 0.f;
    for (int c = 0; c < 128; ++c) { float v = hn[lane * 128 + c]; s += v; ss += v * v; }
    float mean = s * (1.f / 128.f);
    float var  = ss * (1.f / 128.f) - mean * mean;
    smean[lane] = mean; srstd[lane] = rsqrtf(var + 1e-5f);
  }
  __syncthreads();
  for (int idx = lane; idx < 2048; idx += 32) {
    int m = idx >> 7, c = idx & 127, node = n0 + m;
    if (node < A.N) {
      float v = (hn[idx] - smean[m]) * srstd[m] * A.normg[c] + A.normb[c];
      size_t off = (size_t)node * 128 + c;
      float res = A.parity ? (A.h[off] + v) : v;
      A.h[off]  = res;
      A.hf[off] = (_Float16)res;
      A.lo[off] = res;
    }
  }
}

// ---------------------------------------------------------------------------
// Layer-attention weighted mean pool: pooled[c] += sum_n sum_l attn_l*lo_l / N
// ---------------------------------------------------------------------------
__global__ void pool_kernel(const float* __restrict__ lo, const float* __restrict__ lattn,
                            float* __restrict__ pooled, int N) {
  float a0 = lattn[0], a1 = lattn[1], a2 = lattn[2], a3 = lattn[3];
  float mx = fmaxf(fmaxf(a0, a1), fmaxf(a2, a3));
  float e0 = __expf(a0 - mx), e1 = __expf(a1 - mx), e2 = __expf(a2 - mx), e3 = __expf(a3 - mx);
  float inv = 1.f / (e0 + e1 + e2 + e3);
  float w0 = e0 * inv, w1 = e1 * inv, w2 = e2 * inv, w3 = e3 * inv;
  int c = threadIdx.x & 127;
  size_t NH = (size_t)N * 128;
  float s = 0.f;
  for (int n = blockIdx.x * 2 + (threadIdx.x >> 7); n < N; n += gridDim.x * 2) {
    size_t off = (size_t)n * 128 + c;
    s += w0 * lo[off] + w1 * lo[NH + off] + w2 * lo[2 * NH + off] + w3 * lo[3 * NH + off];
  }
  atomicAdd(&pooled[c], s * (1.f / (float)N));
}

// ---------------------------------------------------------------------------
// Head: z = LN(lrelu(pre1(pooled))); z = lrelu(pre2(z)); out = out(z). 1 block.
// ---------------------------------------------------------------------------
__global__ void __launch_bounds__(128) head_kernel(
    const float* pooled, const float* w1, const float* b1,
    const float* lng, const float* lnb, const float* w2, const float* b2,
    const float* w3, const float* b3, float* out) {
  __shared__ float t[128], z[128], z2[64], red[2];
  int tid = threadIdx.x;
  t[tid] = pooled[tid];
  __syncthreads();
  float acc = b1[tid];
  for (int k = 0; k < 128; ++k) acc += w1[tid * 128 + k] * t[k];
  z[tid] = lrelu(acc);
  __syncthreads();
  if (tid == 0) {
    float s = 0.f, ss = 0.f;
    for (int k = 0; k < 128; ++k) { s += z[k]; ss += z[k] * z[k]; }
    float mean = s * (1.f / 128.f);
    float var = ss * (1.f / 128.f) - mean * mean;
    red[0] = mean; red[1] = rsqrtf(var + 1e-5f);
  }
  __syncthreads();
  z[tid] = (z[tid] - red[0]) * red[1] * lng[tid] + lnb[tid];
  __syncthreads();
  if (tid < 64) {
    float a = b2[tid];
    for (int k = 0; k < 128; ++k) a += w2[tid * 128 + k] * z[k];
    z2[tid] = lrelu(a);
  }
  __syncthreads();
  if (tid < 34) {
    float a = b3[tid];
    for (int k = 0; k < 64; ++k) a += w3[tid * 64 + k] * z2[k];
    out[tid] = a;
  }
}

// ---------------------------------------------------------------------------
// Host-side orchestration.
// Param leaves follow JAX sorted-dict flatten order:
//   x(0), edge_index(1), edge_attr(2),
//   emb: comb.b(3) comb.w(4) lin.b(5) lin.w(6) pow.b(7) pow.w(8),
//   layer_attention(9),
//   per layer l (base 10+25l):
//     0 att.b 1 att.w 2 gate.b 3 gate.w 4 m1_1.b 5 m1_1.w 6 m1_2.b 7 m1_2.w
//     8 m1_ln_b 9 m1_ln_g 10 m2_1.b 11 m2_1.w 12 m2_2.b 13 m2_2.w
//     14 m2_ln_b 15 m2_ln_g 16 norm_b 17 norm_g 18 scale
//     19 u1.b 20 u1.w 21 u1_ln_b 22 u1_ln_g 23 u2.b 24 u2.w
//   out.b(110) out.w(111) pre1.b(112) pre1.w(113) pre2.b(114) pre2.w(115)
//   pre_ln_b(116) pre_ln_g(117)
// ---------------------------------------------------------------------------
extern "C" void kernel_launch(void* const* d_in, const int* in_sizes, int n_in,
                              void* d_out, int out_size, void* d_ws, size_t ws_size,
                              hipStream_t stream) {
  const float* x  = (const float*)d_in[0];
  const int*   ei = (const int*)d_in[1];
  const float* ea = (const float*)d_in[2];
  const int N    = in_sizes[0] / 16;
  const int E    = in_sizes[1] / 2;
  const int Etot = E + N;
  const int L = 4;

  auto F = [&](int i) { return (const float*)d_in[i]; };
  const float* emb_comb_b = F(3); const float* emb_comb_w = F(4);
  const float* emb_lin_b  = F(5); const float* emb_lin_w  = F(6);
  const float* emb_pow_b  = F(7); const float* emb_pow_w  = F(8);
  const float* layer_attn = F(9);
  const float* out_b = F(110); const float* out_w = F(111);
  const float* pre1_b = F(112); const float* pre1_w = F(113);
  const float* pre2_b = F(114); const float* pre2_w = F(115);
  const float* pre_ln_b = F(116); const float* pre_ln_g = F(117);

  // Workspace carve (256-byte aligned regions).
  char* wp = (char*)d_ws;
  auto carve = [&](size_t bytes) -> char* {
    char* r = wp; wp += (bytes + 255) & ~(size_t)255; return r;
  };
  float*     h      = (float*)carve((size_t)N * 128 * 4);
  _Float16*  hf     = (_Float16*)carve((size_t)N * 128 * 2);
  float*     aggr   = (float*)carve((size_t)N * 128 * 4);
  float*     lo     = (float*)carve((size_t)4 * N * 128 * 4);
  float*     pooled = (float*)carve(1024);

  const size_t PK9 = 9 * 8 * 512 * 2, PK8 = 8 * 8 * 512 * 2;
  const size_t PK4 = 4 * 8 * 512 * 2, PK1 = 1 * 8 * 512 * 2;
  _Float16* pk_lin  = (_Float16*)carve(PK1);
  _Float16* pk_pow  = (_Float16*)carve(PK1);
  _Float16* pk_comb = (_Float16*)carve(PK8);
  _Float16 *pk11[4], *pk21[4], *pk12[4], *pk22[4], *pkgate[4], *pku1[4], *pku2[4];
  for (int l = 0; l < L; ++l) {
    pk11[l]   = (_Float16*)carve(PK9);
    pk21[l]   = (_Float16*)carve(PK9);
    pk12[l]   = (_Float16*)carve(PK4);
    pk22[l]   = (_Float16*)carve(PK4);
    pkgate[l] = (_Float16*)carve(PK8);
    pku1[l]   = (_Float16*)carve(PK8);
    pku2[l]   = (_Float16*)carve(PK4);
  }

  auto pack = [&](const float* W, int O, int I, int kOff, int KB, _Float16* dst) {
    int total = KB * 8 * 512;
    pack_w<<<dim3((total + 255) / 256), dim3(256), 0, stream>>>(W, O, I, kOff, KB, 8, dst);
  };
  pack(emb_lin_w, 128, 16, 0, 1, pk_lin);
  pack(emb_pow_w, 128, 16, 16, 1, pk_pow);   // pow occupies K=16..31 of [x|x^2]
  pack(emb_comb_w, 128, 256, 0, 8, pk_comb);
  for (int l = 0; l < L; ++l) {
    int base = 10 + 25 * l;
    pack(F(base + 5),  128, 260, 0, 9, pk11[l]);
    pack(F(base + 11), 128, 260, 0, 9, pk21[l]);
    pack(F(base + 7),  128, 128, 0, 4, pk12[l]);
    pack(F(base + 13), 128, 128, 0, 4, pk22[l]);
    pack(F(base + 3),  128, 256, 0, 8, pkgate[l]);
    pack(F(base + 20), 128, 256, 0, 8, pku1[l]);
    pack(F(base + 24), 128, 128, 0, 4, pku2[l]);
  }

  // Embedding.
  const int ntiles  = (N + 15) / 16;
  const int nblocks = (ntiles + 3) / 4;
  {
    EmbArgs A{ x, h, hf, pk_lin, pk_pow, pk_comb, emb_lin_b, emb_pow_b, emb_comb_b, N };
    embed_kernel<<<dim3(nblocks), dim3(128), 4 * EMB_WB, stream>>>(A);
  }

  // Layers.
  const int etiles  = (Etot + 15) / 16;
  const int eblocks = (etiles + 3) / 4;
  for (int l = 0; l < L; ++l) {
    int base = 10 + 25 * l;
    int act = (l & 1);  // gelu on odd layers, lrelu on even
    hipMemsetAsync(aggr, 0, (size_t)N * 128 * 4, stream);
    {
      EdgeArgs A;
      A.hf = hf; A.ei = ei; A.ea = ea;
      A.pk11 = pk11[l]; A.pk21 = pk21[l]; A.pk12 = pk12[l]; A.pk22 = pk22[l];
      A.b11 = F(base + 4);  A.b21 = F(base + 10);
      A.b12 = F(base + 6);  A.b22 = F(base + 12);
      A.ln1g = F(base + 9); A.ln1b = F(base + 8);
      A.ln2g = F(base + 15); A.ln2b = F(base + 14);
      A.attw = F(base + 1); A.attb = F(base + 0); A.scale = F(base + 18);
      A.aggr = aggr; A.E = E; A.Etot = Etot; A.act = act;
      edge_msg_kernel<<<dim3(eblocks), dim3(128), 4 * EDGE_WB, stream>>>(A);
    }
    {
      NodeArgs A;
      A.aggr = aggr; A.h = h; A.hf = hf; A.lo = lo + (size_t)l * N * 128;
      A.pkgate = pkgate[l]; A.pku1 = pku1[l]; A.pku2 = pku2[l];
      A.gate_b = F(base + 2); A.u1_b = F(base + 19); A.u2_b = F(base + 23);
      A.u1lng = F(base + 22); A.u1lnb = F(base + 21);
      A.normg = F(base + 17); A.normb = F(base + 16);
      A.N = N; A.act = act; A.parity = (l & 1);
      node_update_kernel<<<dim3(nblocks), dim3(128), 4 * NODE_WB, stream>>>(A);
    }
  }

  // Readout.
  hipMemsetAsync(pooled, 0, 1024, stream);
  pool_kernel<<<dim3(512), dim3(256), 0, stream>>>(lo, layer_attn, pooled, N);
  head_kernel<<<dim3(1), dim3(128), 0, stream>>>(
      pooled, pre1_w, pre1_b, pre_ln_g, pre_ln_b, pre2_w, pre2_b,
      out_w, out_b, (float*)d_out);
}
